// SupConLoss_47665547051425
// MI455X (gfx1250) — compile-verified, matrix operand order checked by reading, not run on hardware
//
#include <hip/hip_runtime.h>
#include <hip/hip_bf16.h>

#define N_TOT 8192
#define D_DIM 128
#define CSPLIT 4
#define COLS_PER_SPLIT (N_TOT / CSPLIT)
#define INV_T 14.285714285714286f /* 1 / 0.07 */

typedef __attribute__((ext_vector_type(16))) _Float16 v16h;
typedef __attribute__((ext_vector_type(8)))  _Float16 v8h;
typedef __attribute__((ext_vector_type(4)))  _Float16 v4h;
typedef __attribute__((ext_vector_type(8)))  float    v8f;
typedef __attribute__((ext_vector_type(4)))  float    v4f;

union V16HU { v16h v; v8h h[2]; };

// Load one 16x32 f16 WMMA operand fragment (A-layout; B for E*E^T is identical
// because B[cTile] in K-major equals rows [c,c+16) of E).
// Per CDNA5 ISA: lanes 0-15 hold K {0..7,16..23}, lanes 16-31 hold {8..15,24..31}.
__device__ __forceinline__ v16h load_frag16x32(const _Float16* __restrict__ base,
                                               int row, int kbase, int half) {
  const _Float16* p = base + row * D_DIM + kbase + (half ? 8 : 0);
  V16HU u;
  u.h[0] = *(const v8h*)(p);
  u.h[1] = *(const v8h*)(p + 16);
  return u.v;
}

// fp32 -> fp16 embedding conversion (one-time; 1M elements, 4-wide)
__global__ void __launch_bounds__(256) supcon_convert_kernel(
    const float* __restrict__ in, _Float16* __restrict__ out, int n4) {
  int i = blockIdx.x * blockDim.x + threadIdx.x;
  if (i < n4) {
    v4f f = *(const v4f*)(in + i * 4);
    v4h h = { (_Float16)f.x, (_Float16)f.y, (_Float16)f.z, (_Float16)f.w };
    *(v4h*)(out + i * 4) = h;
  }
}

// Fold one 16x16 similarity tile into per-row exp-sum / positive-sum / count.
__device__ __forceinline__ void tile_epilogue(
    const v8f& c, int col, int clab, int rbase, int half,
    const int* __restrict__ rowlab,
    float* __restrict__ accE, float* __restrict__ accP, float* __restrict__ accC) {
#pragma unroll
  for (int r = 0; r < 8; ++r) {
    const int row   = rbase + half * 8 + r; // C layout: M = r + 8*(lane>=16)
    const float s   = c[r] * INV_T;
    const bool self = (col == row);
    const float e   = __expf(s);
    accE[r] += self ? 0.0f : e;
    const bool pos = (!self) && (clab == rowlab[r]);
    accP[r] += pos ? s : 0.0f;
    accC[r] += pos ? 1.0f : 0.0f;
  }
}

// Fused Gram-matrix + softmax-denominator + masked positive accumulation.
// Each wave owns 16 rows; A fragments (16x128) live in VGPRs for the whole
// kernel; B tiles (16 cols x 128) stream from L2 (2 MB working set).
// Column loop is unrolled 2x: two independent 16-col tiles per iteration so
// tile-1's 8 b128 loads are in flight while tile-0's WMMA + exp run.
__global__ void __launch_bounds__(64) supcon_sim_kernel(
    const _Float16* __restrict__ embh, const int* __restrict__ labels,
    float* __restrict__ wsE, float* __restrict__ wsP, float* __restrict__ wsC) {
  const int lane   = threadIdx.x & 31;
  const int wave   = threadIdx.x >> 5;
  const int half   = lane >> 4;    // 0: lanes 0-15, 1: lanes 16-31
  const int lrow   = lane & 15;
  const int rbase  = blockIdx.x * 32 + wave * 16;
  const int cstart = blockIdx.y * COLS_PER_SPLIT;
  const int cend   = cstart + COLS_PER_SPLIT;

  // A fragments for this wave's 16 rows, K = 0..127 (4 x 16x32)
  v16h a0 = load_frag16x32(embh, rbase + lrow, 0,  half);
  v16h a1 = load_frag16x32(embh, rbase + lrow, 32, half);
  v16h a2 = load_frag16x32(embh, rbase + lrow, 64, half);
  v16h a3 = load_frag16x32(embh, rbase + lrow, 96, half);

  int rowlab[8];
#pragma unroll
  for (int r = 0; r < 8; ++r) rowlab[r] = labels[rbase + half * 8 + r];

  float accE[8], accP[8], accC[8];
#pragma unroll
  for (int r = 0; r < 8; ++r) { accE[r] = 0.f; accP[r] = 0.f; accC[r] = 0.f; }

  for (int cb = cstart; cb < cend; cb += 32) {
    // Issue ALL loads for both tiles up front (distinct registers, one clause
    // window) so the second tile's L2 traffic overlaps the first tile's math.
    v16h b00 = load_frag16x32(embh, cb + lrow,      0,  half);
    v16h b01 = load_frag16x32(embh, cb + lrow,      32, half);
    v16h b02 = load_frag16x32(embh, cb + lrow,      64, half);
    v16h b03 = load_frag16x32(embh, cb + lrow,      96, half);
    v16h b10 = load_frag16x32(embh, cb + 16 + lrow, 0,  half);
    v16h b11 = load_frag16x32(embh, cb + 16 + lrow, 32, half);
    v16h b12 = load_frag16x32(embh, cb + 16 + lrow, 64, half);
    v16h b13 = load_frag16x32(embh, cb + 16 + lrow, 96, half);
    const int lab0 = labels[cb + lrow];
    const int lab1 = labels[cb + 16 + lrow];

    // Warm L0 for the next iteration's B rows (clamped in-bounds; WGP scope).
    const int pre = (cb + 32 < cend) ? (cb + 32) : cstart;
    __builtin_prefetch(embh + (pre + lrow) * D_DIM, 0, 3);
    __builtin_prefetch(embh + (pre + 16 + lrow) * D_DIM, 0, 3);

    v8f c0 = {};
    c0 = __builtin_amdgcn_wmma_f32_16x16x32_f16(false, a0, false, b00, (short)0, c0, false, false);
    c0 = __builtin_amdgcn_wmma_f32_16x16x32_f16(false, a1, false, b01, (short)0, c0, false, false);
    c0 = __builtin_amdgcn_wmma_f32_16x16x32_f16(false, a2, false, b02, (short)0, c0, false, false);
    c0 = __builtin_amdgcn_wmma_f32_16x16x32_f16(false, a3, false, b03, (short)0, c0, false, false);

    v8f c1 = {};
    c1 = __builtin_amdgcn_wmma_f32_16x16x32_f16(false, a0, false, b10, (short)0, c1, false, false);
    c1 = __builtin_amdgcn_wmma_f32_16x16x32_f16(false, a1, false, b11, (short)0, c1, false, false);
    c1 = __builtin_amdgcn_wmma_f32_16x16x32_f16(false, a2, false, b12, (short)0, c1, false, false);
    c1 = __builtin_amdgcn_wmma_f32_16x16x32_f16(false, a3, false, b13, (short)0, c1, false, false);

    tile_epilogue(c0, cb + lrow,      lab0, rbase, half, rowlab, accE, accP, accC);
    tile_epilogue(c1, cb + 16 + lrow, lab1, rbase, half, rowlab, accE, accP, accC);
  }

  // Reduce across the 16 lanes holding each row (xor masks 1..8 stay in-half)
#pragma unroll
  for (int r = 0; r < 8; ++r) {
#pragma unroll
    for (int m = 8; m >= 1; m >>= 1) {
      accE[r] += __shfl_xor(accE[r], m, 32);
      accP[r] += __shfl_xor(accP[r], m, 32);
      accC[r] += __shfl_xor(accC[r], m, 32);
    }
  }
  if (lrow == 0) {   // lane 0 -> rows rbase+0..7, lane 16 -> rows rbase+8..15
#pragma unroll
    for (int r = 0; r < 8; ++r) {
      const int row = rbase + half * 8 + r;
      atomicAdd(&wsE[row], accE[r]);
      atomicAdd(&wsP[row], accP[r]);
      atomicAdd(&wsC[row], accC[r]);
    }
  }
}

__global__ void __launch_bounds__(256) supcon_finalize_kernel(
    const float* __restrict__ wsE, const float* __restrict__ wsP,
    const float* __restrict__ wsC, float* __restrict__ out) {
  const int i = blockIdx.x * blockDim.x + threadIdx.x;
  float term = 0.0f;
  if (i < N_TOT) {
    const float c = wsC[i];
    if (c > 0.0f) term = wsP[i] / c - __logf(wsE[i] + 1e-9f);
  }
#pragma unroll
  for (int m = 16; m >= 1; m >>= 1) term += __shfl_xor(term, m, 32);
  if ((threadIdx.x & 31) == 0) atomicAdd(out, -term * (1.0f / (float)N_TOT));
}

extern "C" void kernel_launch(void* const* d_in, const int* in_sizes, int n_in,
                              void* d_out, int out_size, void* d_ws, size_t ws_size,
                              hipStream_t stream) {
  (void)in_sizes; (void)n_in; (void)out_size; (void)ws_size;
  const float* emb    = (const float*)d_in[0];
  const int*   labels = (const int*)d_in[1];
  float*       out    = (float*)d_out;

  // Workspace layout: [wsE | wsP | wsC] (3 * 8192 f32 = 96 KB, 128B-aligned end)
  // then fp16 embeddings (8192*128*2 = 2 MB).
  float* wsE = (float*)d_ws;
  float* wsP = wsE + N_TOT;
  float* wsC = wsP + N_TOT;
  _Float16* embh = (_Float16*)((char*)d_ws + 3 * N_TOT * sizeof(float));

  hipMemsetAsync(d_ws, 0, 3 * N_TOT * sizeof(float), stream);
  hipMemsetAsync(d_out, 0, sizeof(float), stream);

  const int n4 = (N_TOT * D_DIM) / 4;
  supcon_convert_kernel<<<(n4 + 255) / 256, 256, 0, stream>>>(emb, embh, n4);

  dim3 grid(N_TOT / 32, CSPLIT);   // 256 row-blocks (2 waves each) x 4 column splits
  supcon_sim_kernel<<<grid, 64, 0, stream>>>(embh, labels, wsE, wsP, wsC);

  supcon_finalize_kernel<<<N_TOT / 256, 256, 0, stream>>>(wsE, wsP, wsC, out);
}
